// CurvSelfAttention_81784767250782
// MI455X (gfx1250) — matrix operand order repaired and checked by reading, not compiled, not run on hardware
//
#include <hip/hip_runtime.h>
#include <hip/hip_bf16.h>

typedef __bf16 bf16_t;
typedef __bf16 v16bf __attribute__((ext_vector_type(16)));
typedef __bf16 v8bf  __attribute__((ext_vector_type(8)));
typedef float  v8f   __attribute__((ext_vector_type(8)));

#define DMODEL 1024
#define SEQ    2048
#define NHEAD  16
#define HDIM   64

// ---------------------------------------------------------------------------
// WMMA helpers (CDNA5 wave32, 16x16x32 bf16 -> f32 accum)
// ---------------------------------------------------------------------------
__device__ __forceinline__ v8f wmma_bf16(v16bf a, v16bf b, v8f c) {
  return __builtin_amdgcn_wmma_f32_16x16x32_bf16(false, a, false, b, (short)0, c,
                                                 false, false);
}

// A fragment (16 rows x 32 K, bf16) from row-major [row, k] storage (global or LDS).
// Lane L: row = row0 + (L&15); elems e -> K = k0 + 16*hi + e (16 contiguous)
__device__ __forceinline__ v16bf load_a_frag(const bf16_t* base, int ld, int row0, int k0) {
  const int lane = threadIdx.x & 31;
  const int m = lane & 15, hi = lane >> 4;
  const bf16_t* p = base + (size_t)(row0 + m) * ld + k0 + 16 * hi;
  v8bf lo = *(const v8bf*)p;
  v8bf hh = *(const v8bf*)(p + 8);
  return __builtin_shufflevector(lo, hh, 0,1,2,3,4,5,6,7,8,9,10,11,12,13,14,15);
}

// B fragment (32 K x 16 cols, bf16) from N-major storage: element(k,n) at base[n*ld + k].
// Lane L: n = n0 + (L&15); elems e -> K = k0 + 16*hi + e (16 contiguous)
__device__ __forceinline__ v16bf load_b_frag(const bf16_t* base, int ld, int n0, int k0) {
  const int lane = threadIdx.x & 31;
  const int n = lane & 15, hi = lane >> 4;
  const bf16_t* p = base + (size_t)(n0 + n) * ld + k0 + 16 * hi;
  v8bf lo = *(const v8bf*)p;
  v8bf hh = *(const v8bf*)(p + 8);
  return __builtin_shufflevector(lo, hh, 0,1,2,3,4,5,6,7,8,9,10,11,12,13,14,15);
}

// ---------------------------------------------------------------------------
// CDNA5 async global->LDS (tracked by ASYNCcnt); 16 bytes per lane.
// VDST = LDS byte offset VGPR, VADDR = 64-bit global address.
// ---------------------------------------------------------------------------
__device__ __forceinline__ unsigned lds_off(const void* p) {
  // low 32 bits of the shared-aperture flat address == LDS byte offset
  return (unsigned)(size_t)p;
}
__device__ __forceinline__ void async_b128(unsigned lds, const void* gaddr) {
  asm volatile("global_load_async_to_lds_b128 %0, %1, off"
               :: "v"(lds), "v"(gaddr)
               : "memory");
}
__device__ __forceinline__ void wait_async0() {
  asm volatile("s_wait_asynccnt 0x0" ::: "memory");
}

// ---------------------------------------------------------------------------
// Prep: fp32 -> bf16 elementwise
// ---------------------------------------------------------------------------
__global__ __launch_bounds__(256) void cvt_f32_bf16(const float* __restrict__ x,
                                                    bf16_t* __restrict__ y, int n) {
  int i = blockIdx.x * 256 + threadIdx.x;
  if (i < n) y[i] = (bf16_t)x[i];
}

// Prep: transpose+convert the 4 weight matrices: WT[z][c*1024+k] = bf16(W[k*1024+c])
__global__ __launch_bounds__(256) void transpose_cvt(const float* __restrict__ Wq,
                                                     const float* __restrict__ Wk,
                                                     const float* __restrict__ Wv,
                                                     const float* __restrict__ Ws,
                                                     bf16_t* __restrict__ dst) {
  __shared__ float tile[32][33];
  const float* src = (blockIdx.z == 0) ? Wq : (blockIdx.z == 1) ? Wk
                    : (blockIdx.z == 2) ? Wv : Ws;
  bf16_t* out = dst + (size_t)blockIdx.z * DMODEL * DMODEL;
  int k0 = blockIdx.x * 32, c0 = blockIdx.y * 32;
  int tx = threadIdx.x, ty = threadIdx.y;  // 32 x 8
#pragma unroll
  for (int i = 0; i < 4; ++i)
    tile[ty + 8 * i][tx] = src[(size_t)(k0 + ty + 8 * i) * DMODEL + c0 + tx];
  __syncthreads();
#pragma unroll
  for (int i = 0; i < 4; ++i)
    out[(size_t)(c0 + ty + 8 * i) * DMODEL + k0 + tx] = (bf16_t)tile[tx][ty + 8 * i];
}

// ---------------------------------------------------------------------------
// Projection GEMM: D[t,c] = sum_k X[t,k]*W[k,c] (+bias, per-mode epilogue)
// One wave computes a 32x32 output tile (2x2 fragment reuse -> 128 B/output).
//   mode 0: of32[t,c] = sigmoid(v)+0.5                (curvature scale S)
//   mode 1: obf[t,c]  = bf16(v * scale[t,c])          (Qs / Ks)
//   mode 2: obf[((b*16+h)*64+d)*2048+s] = bf16(v)     (V transposed [B,H,D,S])
// ---------------------------------------------------------------------------
__global__ __launch_bounds__(256) void proj_kernel(const bf16_t* __restrict__ X,
                                                   const bf16_t* __restrict__ Wt,
                                                   const float* __restrict__ bias,
                                                   const float* __restrict__ scale,
                                                   bf16_t* __restrict__ obf,
                                                   float* __restrict__ of32,
                                                   int mode) {
  const int wid = blockIdx.x * 8 + (threadIdx.x >> 5);
  const int nt = wid & 31;   // 32 column super-tiles (32 cols each)
  const int mt = wid >> 5;   // 128 row super-tiles (32 rows each)
  const int lane = threadIdx.x & 31;
  const int hi = lane >> 4, ln = lane & 15;

  v8f acc[2][2] = {};
  for (int k0 = 0; k0 < DMODEL; k0 += 32) {
    v16bf a0 = load_a_frag(X, DMODEL, mt * 32, k0);
    v16bf a1 = load_a_frag(X, DMODEL, mt * 32 + 16, k0);
    v16bf b0 = load_b_frag(Wt, DMODEL, nt * 32, k0);
    v16bf b1 = load_b_frag(Wt, DMODEL, nt * 32 + 16, k0);
    acc[0][0] = wmma_bf16(a0, b0, acc[0][0]);
    acc[0][1] = wmma_bf16(a0, b1, acc[0][1]);
    acc[1][0] = wmma_bf16(a1, b0, acc[1][0]);
    acc[1][1] = wmma_bf16(a1, b1, acc[1][1]);
  }

#pragma unroll
  for (int mi = 0; mi < 2; ++mi)
#pragma unroll
    for (int ni = 0; ni < 2; ++ni) {
      const int c = nt * 32 + ni * 16 + ln;
      const float bb = bias[c];
#pragma unroll
      for (int r = 0; r < 8; ++r) {
        int t = mt * 32 + mi * 16 + 8 * hi + r;  // token (C layout: m = 8*hi + r)
        float v = acc[mi][ni][r] + bb;
        if (mode == 0) {
          of32[(size_t)t * DMODEL + c] =
              1.0f / (1.0f + exp2f(-v * 1.4426950408889634f)) + 0.5f;
        } else if (mode == 1) {
          obf[(size_t)t * DMODEL + c] = (bf16_t)(v * scale[(size_t)t * DMODEL + c]);
        } else {
          int b = t >> 11, s = t & (SEQ - 1);
          int h = c >> 6, d = c & 63;
          obf[((size_t)(b * NHEAD + h) * HDIM + d) * SEQ + s] = (bf16_t)v;
        }
      }
    }
}

// ---------------------------------------------------------------------------
// Flash attention, scoresT formulation, K/V staged in LDS via async DMA.
// Block = 4 waves = 64 queries; grid (SEQ/64, NHEAD, B).
// Key loop: 32-key chunks, double-buffered LDS:
//   Kl[buf][32 keys][64 d], Vl[buf][64 d][32 keys]  (4 KB each, 16 KB total)
// Per iteration: wait_asynccnt(0) -> barrier -> issue next chunk DMA ->
// compute current chunk (8 WMMAs + online softmax) from LDS.
// ---------------------------------------------------------------------------
__device__ __forceinline__ void stage_chunk(bf16_t* Kd, bf16_t* Vd,
                                            const bf16_t* Kb, const bf16_t* Vb,
                                            int kc, int tid) {
  // K chunk: 64 keys x 32 d = 256 x 16B; 128 threads x 2 issues
#pragma unroll
  for (int i = 0; i < 2; ++i) {
    int id = tid + i * 128;
    int r = id >> 2, p = (id & 3) * 8;
    async_b128(lds_off(&Kd[r * HDIM + p]), Kb + (size_t)(kc + r) * DMODEL + p);
  }
  // V chunk: 64 d x 32 keys = 256 x 16B
#pragma unroll
  for (int i = 0; i < 2; ++i) {
    int id = tid + i * 128;
    int d = id >> 2, p = (id & 3) * 8;
    async_b128(lds_off(&Vd[d * 32 + p]), Vb + (size_t)d * SEQ + kc + p);
  }
}

__global__ __launch_bounds__(128) void attn_kernel(const bf16_t* __restrict__ Qs,
                                                   const bf16_t* __restrict__ Ks,
                                                   const bf16_t* __restrict__ Vt,
                                                   float* __restrict__ out) {
  __shared__ bf16_t Kl[2][32 * HDIM];
  __shared__ bf16_t Vl[2][HDIM * 32];

  const int tid = threadIdx.x;
  const int wave = tid >> 5;
  const int lane = tid & 31;
  const int hi = lane >> 4;
  const int q0 = blockIdx.x * 64 + wave * 16;
  const int h = blockIdx.y;
  const int b = blockIdx.z;

  const bf16_t* Qb = Qs + (size_t)b * SEQ * DMODEL + h * HDIM;
  const bf16_t* Kb = Ks + (size_t)b * SEQ * DMODEL + h * HDIM;
  const bf16_t* Vb = Vt + (size_t)(b * NHEAD + h) * HDIM * SEQ;

  // Qs^T B-fragments over d, held in registers for the whole key loop
  v16bf bq0 = load_b_frag(Qb, DMODEL, q0, 0);
  v16bf bq1 = load_b_frag(Qb, DMODEL, q0, 32);

  stage_chunk(Kl[0], Vl[0], Kb, Vb, 0, tid);

  v8f acc[4] = {};               // ctxT, 4 d-tiles of 16
  float m_cur = -1e30f, l_cur = 0.0f;
  const float SCALE = 0.125f;    // 1/sqrt(64)
  const float LOG2E = 1.4426950408889634f;
  int cur = 0;

  for (int kc = 0; kc < SEQ; kc += 32) {
    wait_async0();       // my 4 outstanding DMAs for chunk kc are done
    __syncthreads();     // all waves' DMAs landed; prev buffer free for reuse
    if (kc + 32 < SEQ)   // overlap next chunk's DMA with this chunk's compute
      stage_chunk(Kl[cur ^ 1], Vl[cur ^ 1], Kb, Vb, kc + 32, tid);

    const bf16_t* Kc = Kl[cur];
    const bf16_t* Vc = Vl[cur];

    // two scoresT tiles (16 keys x 16 q each) from LDS
    v8f st0 = {}, st1 = {};
    {
      v16bf ak = load_a_frag(Kc, HDIM, 0, 0);
      st0 = wmma_bf16(ak, bq0, st0);
      ak = load_a_frag(Kc, HDIM, 0, 32);
      st0 = wmma_bf16(ak, bq1, st0);
      ak = load_a_frag(Kc, HDIM, 16, 0);
      st1 = wmma_bf16(ak, bq0, st1);
      ak = load_a_frag(Kc, HDIM, 16, 32);
      st1 = wmma_bf16(ak, bq1, st1);
    }
    // online softmax (per lane: fixed q, 16 key values; pair-lane reduce via xor 16)
    float p0[8], p1[8], mx = -1e30f;
#pragma unroll
    for (int r = 0; r < 8; ++r) {
      p0[r] = st0[r] * SCALE;
      p1[r] = st1[r] * SCALE;
      mx = fmaxf(mx, fmaxf(p0[r], p1[r]));
    }
    mx = fmaxf(mx, __shfl_xor(mx, 16, 32));
    const float m_new = fmaxf(m_cur, mx);
    const float alpha = exp2f((m_cur - m_new) * LOG2E);
    float rsum = 0.0f;
#pragma unroll
    for (int r = 0; r < 8; ++r) {
      p0[r] = exp2f((p0[r] - m_new) * LOG2E);
      p1[r] = exp2f((p1[r] - m_new) * LOG2E);
      rsum += p0[r] + p1[r];
    }
    rsum += __shfl_xor(rsum, 16, 32);
    l_cur = l_cur * alpha + rsum;
    m_cur = m_new;
#pragma unroll
    for (int d = 0; d < 4; ++d)
#pragma unroll
      for (int r = 0; r < 8; ++r) acc[d][r] *= alpha;

    // Rebuild probsT as a B fragment (32 keys x 16 q): lane needs tile=hi,
    // elem e<8 from lane-half 0 (m=e), e>=8 from lane-half 1 (m=8+(e-8)).
    v16bf pb;
#pragma unroll
    for (int r = 0; r < 8; ++r) {
      float px0 = __shfl_xor(p0[r], 16, 32);
      float px1 = __shfl_xor(p1[r], 16, 32);
      float lo = hi ? px1 : p0[r];   // tile hi, m = r       (from half 0)
      float hh = hi ? p1[r] : px0;   // tile hi, m = 8 + r   (from half 1)
      pb[r] = (bf16_t)lo;
      pb[r + 8] = (bf16_t)hh;
    }
    // ctxT += Vt(d-tile, key-chunk) x probsT, V from LDS
#pragma unroll
    for (int d = 0; d < 4; ++d) {
      v16bf av = load_a_frag(Vc, 32, d * 16, 0);
      acc[d] = wmma_bf16(av, pb, acc[d]);
    }
    cur ^= 1;
  }

  // epilogue: out[b, s=q, h*64+d] = ctxT[d,q] / l
  const float inv_l = 1.0f / l_cur;
  const int q = q0 + (lane & 15);
#pragma unroll
  for (int d = 0; d < 4; ++d)
#pragma unroll
    for (int r = 0; r < 8; ++r) {
      int dd = d * 16 + 8 * hi + r;
      out[((size_t)b * SEQ + q) * DMODEL + h * HDIM + dd] = acc[d][r] * inv_l;
    }
}

// ---------------------------------------------------------------------------
// Host launcher
// ---------------------------------------------------------------------------
extern "C" void kernel_launch(void* const* d_in, const int* in_sizes, int n_in,
                              void* d_out, int out_size, void* d_ws, size_t ws_size,
                              hipStream_t stream) {
  const float* hs = (const float*)d_in[0];
  const float* Wq = (const float*)d_in[1];
  const float* bq = (const float*)d_in[2];
  const float* Wk = (const float*)d_in[3];
  const float* bk = (const float*)d_in[4];
  const float* Wv = (const float*)d_in[5];
  const float* bv = (const float*)d_in[6];
  const float* Ws = (const float*)d_in[7];
  const float* bs = (const float*)d_in[8];

  char* ws = (char*)d_ws;
  bf16_t* XB = (bf16_t*)(ws + (0ull << 20));   // 8 MiB  X bf16 [4096,1024]
  bf16_t* WT = (bf16_t*)(ws + (8ull << 20));   // 8 MiB  4x Wt bf16 [c,k]
  float*  SS = (float*)(ws + (16ull << 20));   // 16 MiB scale fp32 [4096,1024]
  bf16_t* QS = (bf16_t*)(ws + (32ull << 20));  // 8 MiB  Qs bf16 [t,c]
  bf16_t* KS = (bf16_t*)(ws + (40ull << 20));  // 8 MiB  Ks bf16 [t,c]
  bf16_t* VT = (bf16_t*)(ws + (48ull << 20));  // 8 MiB  V  bf16 [B,H,D,S]

  const int NTOK = 2 * SEQ;  // 4096
  const size_t WSZ = (size_t)DMODEL * DMODEL;

  cvt_f32_bf16<<<NTOK * DMODEL / 256, 256, 0, stream>>>(hs, XB, NTOK * DMODEL);
  transpose_cvt<<<dim3(DMODEL / 32, DMODEL / 32, 4), dim3(32, 8), 0, stream>>>(
      Wq, Wk, Wv, Ws, WT);

  // S first (needed by Q/K epilogues), then Qs, Ks, V(transposed)
  proj_kernel<<<512, 256, 0, stream>>>(XB, WT + 3 * WSZ, bs, nullptr, nullptr, SS, 0);
  proj_kernel<<<512, 256, 0, stream>>>(XB, WT + 0 * WSZ, bq, SS, QS, nullptr, 1);
  proj_kernel<<<512, 256, 0, stream>>>(XB, WT + 1 * WSZ, bk, SS, KS, nullptr, 1);
  proj_kernel<<<512, 256, 0, stream>>>(XB, WT + 2 * WSZ, bv, nullptr, VT, nullptr, 2);

  attn_kernel<<<dim3(SEQ / 64, NHEAD, 2), 128, 0, stream>>>(QS, KS, VT, (float*)d_out);
}
